// ABlock_78563541778940
// MI455X (gfx1250) — matrix operand order repaired and checked by reading, not compiled, not run on hardware
//
#include <hip/hip_runtime.h>
#include <math.h>

// ---------------- problem constants ----------------
constexpr int B_   = 16;
constexpr int C_   = 768;
constexpr int T_   = 9;
constexpr int S_   = T_ * T_ * T_;     // 729 tokens
constexpr int BS_  = B_ * S_;          // 11664
constexpr int H_   = 8;
constexpr int Dh_  = C_ / H_;          // 96
constexpr int E_   = 3;
constexpr int HID_ = C_ * 4;           // 3072
constexpr int NQKV_ = 3 * C_;          // 2304
constexpr int SP_  = 736;              // S padded to multiple of 32

typedef __attribute__((ext_vector_type(16))) _Float16 v16h;
typedef __attribute__((ext_vector_type(8)))  _Float16 v8h;
typedef __attribute__((ext_vector_type(8)))  float    v8f;

// generic->LDS(AS3) 32-bit address for async-copy VDST operand
__device__ __forceinline__ unsigned to_lds(const void* p) {
    return (unsigned)(size_t)(const __attribute__((address_space(3))) void*)p;
}
// 16B async global->LDS copy (GLOBAL_LOAD_ASYNC_TO_LDS_B128, ASYNCcnt-tracked)
__device__ __forceinline__ void async_cp16(unsigned lds, const void* g) {
    asm volatile("global_load_async_to_lds_b128 %0, %1, off"
                 :: "v"(lds), "v"(g) : "memory");
}

// ---------------- generic WMMA GEMM ----------------
// C[M,N] = A[M,K] * Bt[N,K]^T (both f16 row-major, f32 acc).
// Block tile 128x64, K-step 32. 8 waves as 4(m) x 2(n); each wave computes a
// 32x32 tile with 4 WMMA accumulators. Staging uses a double-buffered,
// branch-free software pipeline of global_load_async_to_lds_b128 (no VGPR
// round trip). Staging loads are unconditional (indices clamped: garbage
// rows/cols are row/col-local and never stored).
// Batched over blockIdx.z = b*Hn + h with (b,h) strides.
// EPI: 0 = store f32, 1 = store f16, 2 = bias + exact GELU -> f16,
//      3 = (acc + bias) * gate_score[row,expert] accumulated into f32 out.
template<int EPI>
__global__ __launch_bounds__(256) void gemm_wmma(
    const _Float16* __restrict__ A,  long aSB, long aSH, int lda,
    const _Float16* __restrict__ Bt, long bSB, long bSH, int ldb,
    long cSB, long cSH, int ldc,
    int M, int N, int K, int Hn,
    float* __restrict__ outF, _Float16* __restrict__ outH,
    const float* __restrict__ bias,
    const float* __restrict__ gs, int expert, int accumulate)
{
    constexpr int BLK_M = 128, BLK_N = 64, BLK_K = 32;
    constexpr int LDS_S = 48;   // halves; row stride 96B keeps b128 LDS ops aligned
    __shared__ __align__(32) _Float16 sA[2][BLK_M * LDS_S];   // sA[buf][m][k]
    __shared__ __align__(32) _Float16 sB[2][BLK_N * LDS_S];   // sB[buf][n][k]

    const int z  = blockIdx.z;
    const int zb = z / Hn, zh = z % Hn;
    A  += (long)zb * aSB + (long)zh * aSH;
    Bt += (long)zb * bSB + (long)zh * bSH;
    const long cOff = (long)zb * cSB + (long)zh * cSH;

    const int mbase = blockIdx.y * BLK_M;
    const int nbase = blockIdx.x * BLK_N;
    const int t = threadIdx.x;
    const int lane = t & 31, wave = t >> 5;
    const int wm = wave >> 1, wn = wave & 1;     // 4 x 2 wave grid
    const int hx = lane >> 4, lr = lane & 15;    // half-wave, lane-in-half

    // staging coordinates (8 halves = 16B per thread-slot)
    const int cu = t & 3;                  // which 8-half chunk of the 32-half row
    const int ra = t >> 2;                 // row slot (0..63), +64 on 2nd A pass
    // clamped global rows: unconditional loads; clamped rows produce
    // row/col-local garbage that is masked at the store stage
    int mg0 = mbase + ra;        if (mg0 >= M) mg0 = M - 1;
    int mg1 = mbase + ra + 64;   if (mg1 >= M) mg1 = M - 1;
    int ng  = nbase + ra;        if (ng  >= N) ng  = N - 1;
    const _Float16* gA0 = A  + (long)mg0 * lda + cu * 8;
    const _Float16* gA1 = A  + (long)mg1 * lda + cu * 8;
    const _Float16* gB  = Bt + (long)ng  * ldb + cu * 8;

    // per-buffer LDS destinations for this thread's three 16B async copies
    unsigned dA0[2], dA1[2], dB0[2];
    #pragma unroll
    for (int bi = 0; bi < 2; ++bi) {
        dA0[bi] = to_lds(&sA[bi][ ra       * LDS_S + cu * 8]);
        dA1[bi] = to_lds(&sA[bi][(ra + 64) * LDS_S + cu * 8]);
        dB0[bi] = to_lds(&sB[bi][ ra       * LDS_S + cu * 8]);
    }

    v8f c00 = {}, c01 = {}, c10 = {}, c11 = {};

    // fragment-load + 4x WMMA body for buffer `cur`
    auto compute = [&](int cur) {
        const _Float16* bufA = sA[cur];
        const _Float16* bufB = sB[cur];
        v16h a0, a1, b0, b1;
        // A frag (row=lr, half=hx): halves {8h..8h+7} then {16+8h..23+8h}
        const _Float16* pa0 = &bufA[(wm * 32 + lr) * LDS_S + 8 * hx];
        ((v8h*)&a0)[0] = *(const v8h*)(pa0);
        ((v8h*)&a0)[1] = *(const v8h*)(pa0 + 16);
        const _Float16* pa1 = pa0 + 16 * LDS_S;
        ((v8h*)&a1)[0] = *(const v8h*)(pa1);
        ((v8h*)&a1)[1] = *(const v8h*)(pa1 + 16);
        // B frag (col=lr, half=hx): halves {8h..8h+7} then {16+8h..23+8h}
        const _Float16* pb0 = &bufB[(wn * 32      + lr) * LDS_S + 8 * hx];
        ((v8h*)&b0)[0] = *(const v8h*)(pb0);
        ((v8h*)&b0)[1] = *(const v8h*)(pb0 + 16);
        const _Float16* pb1 = pb0 + 16 * LDS_S;
        ((v8h*)&b1)[0] = *(const v8h*)(pb1);
        ((v8h*)&b1)[1] = *(const v8h*)(pb1 + 16);
        c00 = __builtin_amdgcn_wmma_f32_16x16x32_f16(false, a0, false, b0, (short)0, c00, false, false);
        c01 = __builtin_amdgcn_wmma_f32_16x16x32_f16(false, a0, false, b1, (short)0, c01, false, false);
        c10 = __builtin_amdgcn_wmma_f32_16x16x32_f16(false, a1, false, b0, (short)0, c10, false, false);
        c11 = __builtin_amdgcn_wmma_f32_16x16x32_f16(false, a1, false, b1, (short)0, c11, false, false);
    };

    const int nk = K / BLK_K;
    // prologue: stage tile 0 into buffer 0
    async_cp16(dA0[0], gA0); gA0 += BLK_K;
    async_cp16(dA1[0], gA1); gA1 += BLK_K;
    async_cp16(dB0[0], gB);  gB  += BLK_K;

    // steady state: branch-free issue of tile kt+1 while computing tile kt
    for (int kt = 0; kt < nk - 1; ++kt) {
        const int nxt = (kt + 1) & 1;
        async_cp16(dA0[nxt], gA0); gA0 += BLK_K;
        async_cp16(dA1[nxt], gA1); gA1 += BLK_K;
        async_cp16(dB0[nxt], gB);  gB  += BLK_K;
        // 3 newer copies may remain in flight; tile kt's 3 are complete
        asm volatile("s_wait_asynccnt 0x3" ::: "memory");
        __syncthreads();
        compute(kt & 1);
        __syncthreads();   // all waves done with this buf before it is refilled
    }
    // peeled last tile
    asm volatile("s_wait_asynccnt 0x0" ::: "memory");
    __syncthreads();
    compute((nk - 1) & 1);

    // ---- epilogue: D VGPR r -> row = r + 8*half, col = lane&15 ----
    #pragma unroll
    for (int mi = 0; mi < 2; ++mi) {
        #pragma unroll
        for (int ni = 0; ni < 2; ++ni) {
            const v8f& acc = (mi == 0) ? (ni == 0 ? c00 : c01)
                                       : (ni == 0 ? c10 : c11);
            const int col = nbase + wn * 32 + ni * 16 + lr;
            if (col >= N) continue;
            #pragma unroll
            for (int r = 0; r < 8; ++r) {
                int row = mbase + wm * 32 + mi * 16 + hx * 8 + r;
                if (row >= M) continue;
                float v = acc[r];
                long o = cOff + (long)row * ldc + col;
                if (EPI == 0) {
                    outF[o] = v;
                } else if (EPI == 1) {
                    outH[o] = (_Float16)v;
                } else if (EPI == 2) {
                    v += bias[col];
                    float g = 0.5f * v * (1.0f + erff(v * 0.70710678118654752f));
                    outH[o] = (_Float16)g;
                } else {
                    v += bias[col];
                    float w = gs[(long)row * E_ + expert];
                    float tv = w * v;
                    if (accumulate) outF[o] += tv; else outF[o] = tv;
                }
            }
        }
    }
}

// ---------------- small kernels ----------------
// src f32 [R, Cc] row-major -> dst f16 [Cc, R] row-major, batched over z
__global__ void transpose_f32_f16(const float* __restrict__ src,
                                  _Float16* __restrict__ dst, int R, int Cc) {
    const long base = (long)blockIdx.z * R * Cc;
    long o = (long)blockIdx.x * blockDim.x + threadIdx.x;
    if (o >= (long)R * Cc) return;
    int n = (int)(o / R), r = (int)(o % R);
    dst[base + o] = (_Float16)src[base + (long)r * Cc + n];
}

__global__ void ln_stats(const float* __restrict__ x,
                         float* __restrict__ meanB, float* __restrict__ rstdB) {
    const int b = blockIdx.x;
    const long N = (long)C_ * S_;
    const float* p = x + (long)b * N;
    float s = 0.f, ss = 0.f;
    for (long i = threadIdx.x; i < N; i += blockDim.x) {
        float v = p[i]; s += v; ss += v * v;
    }
    __shared__ float sh1[256], sh2[256];
    sh1[threadIdx.x] = s; sh2[threadIdx.x] = ss;
    __syncthreads();
    for (int o = 128; o > 0; o >>= 1) {
        if (threadIdx.x < o) { sh1[threadIdx.x] += sh1[threadIdx.x + o];
                               sh2[threadIdx.x] += sh2[threadIdx.x + o]; }
        __syncthreads();
    }
    if (threadIdx.x == 0) {
        float mu = sh1[0] / (float)N;
        float var = sh2[0] / (float)N - mu * mu;
        meanB[b] = mu;
        rstdB[b] = rsqrtf(var + 1e-5f);
    }
}

// xn = (x - mu)*rstd*gamma + beta, packed token-major f16 [BS, C]
__global__ void ln_pack(const float* __restrict__ x,
                        const float* __restrict__ gamma, const float* __restrict__ beta,
                        const float* __restrict__ meanB, const float* __restrict__ rstdB,
                        _Float16* __restrict__ tok16) {
    long i = (long)blockIdx.x * blockDim.x + threadIdx.x;
    const long total = (long)BS_ * C_;
    if (i >= total) return;
    int c = (int)(i % C_);
    long m = i / C_;
    int b = (int)(m / S_), s = (int)(m % S_);
    float v = x[((long)b * C_ + c) * S_ + s];
    float xn = (v - meanB[b]) * rstdB[b] * gamma[(long)c * S_ + s] + beta[(long)c * S_ + s];
    tok16[i] = (_Float16)xn;
}

// gather V into zero-padded V^T: vT[b,h,d,sp] (sp 729..735 = 0)
__global__ void pack_vT(const _Float16* __restrict__ qkv, _Float16* __restrict__ vT) {
    const long total = (long)B_ * H_ * Dh_ * SP_;
    long o = (long)blockIdx.x * blockDim.x + threadIdx.x;
    if (o >= total) return;
    int s = (int)(o % SP_);
    long tmp = o / SP_;
    int d = (int)(tmp % Dh_);
    long bh = tmp / Dh_;
    int b = (int)(bh / H_), h = (int)(bh % H_);
    _Float16 v = (_Float16)0.f;
    if (s < S_) v = qkv[((long)b * S_ + s) * NQKV_ + 2 * C_ + (long)h * Dh_ + d];
    vT[o] = v;
}

// softmax over key dim for one (bh, s_q) row; writes padded f16 probs (pad = 0)
__global__ void softmax_rows(const float* __restrict__ scores, _Float16* __restrict__ probs) {
    long row = (long)blockIdx.y * S_ + blockIdx.x;
    const float* src = scores + row * SP_;
    _Float16* dst = probs + row * SP_;
    __shared__ float sh[256];
    float m = -1e30f;
    for (int i = threadIdx.x; i < S_; i += 256) m = fmaxf(m, src[i]);
    sh[threadIdx.x] = m; __syncthreads();
    for (int o = 128; o > 0; o >>= 1) {
        if (threadIdx.x < o) sh[threadIdx.x] = fmaxf(sh[threadIdx.x], sh[threadIdx.x + o]);
        __syncthreads();
    }
    m = sh[0]; __syncthreads();
    float s = 0.f;
    for (int i = threadIdx.x; i < S_; i += 256) s += expf(src[i] - m);
    sh[threadIdx.x] = s; __syncthreads();
    for (int o = 128; o > 0; o >>= 1) {
        if (threadIdx.x < o) sh[threadIdx.x] += sh[threadIdx.x + o];
        __syncthreads();
    }
    float inv = 1.0f / sh[0];
    for (int i = threadIdx.x; i < SP_; i += 256)
        dst[i] = (i < S_) ? (_Float16)(expf(src[i] - m) * inv) : (_Float16)0.0f;
}

// x2 = x + attn_out (token-major f32) and its f16 pack
__global__ void residual_pack(const float* __restrict__ x,
                              const float* __restrict__ attnF,
                              float* __restrict__ x2, _Float16* __restrict__ tok2) {
    long i = (long)blockIdx.x * blockDim.x + threadIdx.x;
    const long total = (long)BS_ * C_;
    if (i >= total) return;
    int c = (int)(i % C_);
    long m = i / C_;
    int b = (int)(m / S_), s = (int)(m % S_);
    float v = x[((long)b * C_ + c) * S_ + s] + attnF[i];
    x2[i] = v;
    tok2[i] = (_Float16)v;
}

// one wave per token: gate logits -> softmax -> top-1 masked gate (mg)
__global__ void gate_mg(const float* __restrict__ x2,
                        const float* __restrict__ wg, const float* __restrict__ bg,
                        float* __restrict__ mg) {
    int wave = threadIdx.x >> 5, lane = threadIdx.x & 31;
    long m = (long)blockIdx.x * 8 + wave;
    if (m >= BS_) return;
    const float* xr = x2 + m * C_;
    float a0 = 0.f, a1 = 0.f, a2 = 0.f;
    for (int c = lane; c < C_; c += 32) {
        float xv = xr[c];
        a0 += xv * wg[c * 3 + 0];
        a1 += xv * wg[c * 3 + 1];
        a2 += xv * wg[c * 3 + 2];
    }
    for (int o = 16; o > 0; o >>= 1) {
        a0 += __shfl_xor(a0, o, 32);
        a1 += __shfl_xor(a1, o, 32);
        a2 += __shfl_xor(a2, o, 32);
    }
    if (lane == 0) {
        a0 += bg[0]; a1 += bg[1]; a2 += bg[2];
        float mx = fmaxf(a0, fmaxf(a1, a2));
        float e0 = expf(a0 - mx), e1 = expf(a1 - mx), e2 = expf(a2 - mx);
        float inv = 1.f / (e0 + e1 + e2);
        float p0 = e0 * inv, p1 = e1 * inv, p2 = e2 * inv;
        int am = 0; float bp = p0;
        if (p1 > bp) { am = 1; bp = p1; }
        if (p2 > bp) { am = 2; bp = p2; }
        mg[m * 3 + 0] = (am == 0) ? p0 : 0.f;
        mg[m * 3 + 1] = (am == 1) ? p1 : 0.f;
        mg[m * 3 + 2] = (am == 2) ? p2 : 0.f;
    }
}

// denom over batch dim (reference sums axis 0), gate_scores = mg/denom*B
__global__ void denom_gs(const float* __restrict__ mg, float* __restrict__ gs) {
    int idx = blockIdx.x * blockDim.x + threadIdx.x;
    if (idx >= S_ * E_) return;
    int s = idx / E_, e = idx % E_;
    float d = 1e-6f;
    for (int b = 0; b < B_; ++b) d += mg[((long)b * S_ + s) * E_ + e];
    float sc = (float)B_ / d;
    for (int b = 0; b < B_; ++b) {
        long i = ((long)b * S_ + s) * E_ + e;
        gs[i] = mg[i] * sc;
    }
}

// out[b,c,s] = x2_tok[m,c] + moe[m,c]
__global__ void final_combine(const float* __restrict__ x2,
                              const float* __restrict__ moe,
                              float* __restrict__ out) {
    long i = (long)blockIdx.x * blockDim.x + threadIdx.x;
    const long total = (long)B_ * C_ * S_;
    if (i >= total) return;
    int s = (int)(i % S_);
    int c = (int)((i / S_) % C_);
    int b = (int)(i / ((long)C_ * S_));
    long m = (long)b * S_ + s;
    out[i] = x2[m * C_ + c] + moe[m * C_ + c];
}

// ---------------- host launcher ----------------
extern "C" void kernel_launch(void* const* d_in, const int* in_sizes, int n_in,
                              void* d_out, int out_size, void* d_ws, size_t ws_size,
                              hipStream_t stream) {
    const float* x      = (const float*)d_in[0];
    const float* gamma  = (const float*)d_in[1];
    const float* beta   = (const float*)d_in[2];
    const float* wqkv   = (const float*)d_in[3];
    const float* wout   = (const float*)d_in[4];
    const float* wgate  = (const float*)d_in[5];
    const float* bgate  = (const float*)d_in[6];
    const float* w1     = (const float*)d_in[7];
    const float* b1     = (const float*)d_in[8];
    const float* w2     = (const float*)d_in[9];
    const float* b2     = (const float*)d_in[10];
    float* out = (float*)d_out;

    // workspace carve-out (256B aligned slices)
    char* ws = (char*)d_ws;
    size_t off = 0;
    auto take = [&](size_t bytes) -> void* {
        void* p = ws + off;
        off += (bytes + 255) & ~(size_t)255;
        return p;
    };
    float*    meanB    = (float*)take(sizeof(float) * B_);
    float*    rstdB    = (float*)take(sizeof(float) * B_);
    _Float16* tok16    = (_Float16*)take((size_t)BS_ * C_ * 2);
    _Float16* wqkvT16  = (_Float16*)take((size_t)NQKV_ * C_ * 2);     // [2304,768]
    _Float16* woutT16  = (_Float16*)take((size_t)C_ * C_ * 2);        // [768,768]
    _Float16* w1T16    = (_Float16*)take((size_t)E_ * HID_ * C_ * 2); // [e][3072,768]
    _Float16* w2T16    = (_Float16*)take((size_t)E_ * C_ * HID_ * 2); // [e][768,3072]
    _Float16* qkv16    = (_Float16*)take((size_t)BS_ * NQKV_ * 2);
    _Float16* vT16     = (_Float16*)take((size_t)B_ * H_ * Dh_ * SP_ * 2);
    float*    scores   = (float*)take((size_t)B_ * H_ * S_ * SP_ * 4);
    _Float16* probs16  = (_Float16*)take((size_t)B_ * H_ * S_ * SP_ * 2);
    _Float16* ctx16    = (_Float16*)take((size_t)BS_ * C_ * 2);
    float*    attnF    = (float*)take((size_t)BS_ * C_ * 4);
    float*    x2buf    = (float*)take((size_t)BS_ * C_ * 4);
    _Float16* tok2_16  = (_Float16*)take((size_t)BS_ * C_ * 2);
    float*    mgbuf    = (float*)take((size_t)BS_ * E_ * 4);
    float*    gsbuf    = (float*)take((size_t)BS_ * E_ * 4);
    _Float16* hmid16   = (_Float16*)take((size_t)BS_ * HID_ * 2);
    float*    moebuf   = (float*)take((size_t)BS_ * C_ * 4);
    (void)ws_size; (void)in_sizes; (void)n_in; (void)out_size;

    // 1) weights -> transposed f16 (one-time)
    transpose_f32_f16<<<dim3((C_ * NQKV_ + 255) / 256, 1, 1), 256, 0, stream>>>(
        wqkv, wqkvT16, C_, NQKV_);
    transpose_f32_f16<<<dim3((C_ * C_ + 255) / 256, 1, 1), 256, 0, stream>>>(
        wout, woutT16, C_, C_);
    transpose_f32_f16<<<dim3((C_ * HID_ + 255) / 256, 1, E_), 256, 0, stream>>>(
        w1, w1T16, C_, HID_);
    transpose_f32_f16<<<dim3((HID_ * C_ + 255) / 256, 1, E_), 256, 0, stream>>>(
        w2, w2T16, HID_, C_);

    // 2) LayerNorm stats + normalize/pack token-major f16
    ln_stats<<<B_, 256, 0, stream>>>(x, meanB, rstdB);
    {
        long total = (long)BS_ * C_;
        ln_pack<<<(unsigned)((total + 255) / 256), 256, 0, stream>>>(
            x, gamma, beta, meanB, rstdB, tok16);
    }

    // 3) qkv = tok16 @ wqkv -> f16  [BS, 2304]
    {
        dim3 g(NQKV_ / 64, (BS_ + 127) / 128, 1);
        gemm_wmma<1><<<g, 256, 0, stream>>>(
            tok16, 0, 0, C_, wqkvT16, 0, 0, C_,
            0, 0, NQKV_, BS_, NQKV_, C_, 1,
            nullptr, qkv16, nullptr, nullptr, 0, 0);
    }

    // 4) V^T gather (zero-padded to 736 key positions)
    {
        long total = (long)B_ * H_ * Dh_ * SP_;
        pack_vT<<<(unsigned)((total + 255) / 256), 256, 0, stream>>>(qkv16, vT16);
    }

    // 5) scores[b,h] = Q @ K^T  (f32, no 1/sqrt(d) per reference)
    {
        dim3 g((S_ + 63) / 64, (S_ + 127) / 128, B_ * H_);
        gemm_wmma<0><<<g, 256, 0, stream>>>(
            qkv16,      (long)S_ * NQKV_, Dh_, NQKV_,     // A  = Q rows
            qkv16 + C_, (long)S_ * NQKV_, Dh_, NQKV_,     // Bt = K rows [S, d]
            (long)H_ * S_ * SP_, (long)S_ * SP_, SP_,
            S_, S_, Dh_, H_,
            scores, nullptr, nullptr, nullptr, 0, 0);
    }

    // 6) row softmax -> padded f16 probs (pad cols zeroed so K=736 is exact)
    {
        dim3 g(S_, B_ * H_);
        softmax_rows<<<g, 256, 0, stream>>>(scores, probs16);
    }

    // 7) ctx[b,h] = probs @ V -> f16 ctx in [BS, C] at column offset h*Dh
    {
        dim3 g((Dh_ + 63) / 64, (S_ + 127) / 128, B_ * H_);
        gemm_wmma<1><<<g, 256, 0, stream>>>(
            probs16, (long)H_ * S_ * SP_, (long)S_ * SP_, SP_,   // A  = probs
            vT16,    (long)H_ * Dh_ * SP_, (long)Dh_ * SP_, SP_, // Bt = V^T [d, Sp]
            (long)S_ * C_, Dh_, C_,
            S_, Dh_, SP_, H_,
            nullptr, ctx16, nullptr, nullptr, 0, 0);
    }

    // 8) attn_out = ctx16 @ w_out -> f32 [BS, C]
    {
        dim3 g(C_ / 64, (BS_ + 127) / 128, 1);
        gemm_wmma<0><<<g, 256, 0, stream>>>(
            ctx16, 0, 0, C_, woutT16, 0, 0, C_,
            0, 0, C_, BS_, C_, C_, 1,
            attnF, nullptr, nullptr, nullptr, 0, 0);
    }

    // 9) residual x2 = x + attn_out; pack f16
    {
        long total = (long)BS_ * C_;
        residual_pack<<<(unsigned)((total + 255) / 256), 256, 0, stream>>>(
            x, attnF, x2buf, tok2_16);
    }

    // 10) gate: top-1 masked softmax, then batch-sum denominator scaling
    gate_mg<<<(BS_ + 7) / 8, 256, 0, stream>>>(x2buf, wgate, bgate, mgbuf);
    denom_gs<<<(S_ * E_ + 255) / 256, 256, 0, stream>>>(mgbuf, gsbuf);

    // 11) MoE experts: up-proj + GELU (f16), down-proj with gated accumulate
    for (int e = 0; e < E_; ++e) {
        {
            dim3 g(HID_ / 64, (BS_ + 127) / 128, 1);
            gemm_wmma<2><<<g, 256, 0, stream>>>(
                tok2_16, 0, 0, C_,
                w1T16 + (size_t)e * HID_ * C_, 0, 0, C_,
                0, 0, HID_, BS_, HID_, C_, 1,
                nullptr, hmid16, b1 + (size_t)e * HID_, nullptr, 0, 0);
        }
        {
            dim3 g(C_ / 64, (BS_ + 127) / 128, 1);
            gemm_wmma<3><<<g, 256, 0, stream>>>(
                hmid16, 0, 0, HID_,
                w2T16 + (size_t)e * C_ * HID_, 0, 0, HID_,
                0, 0, C_, BS_, C_, HID_, 1,
                moebuf, nullptr, b2 + (size_t)e * C_, gsbuf, e, (e > 0) ? 1 : 0);
        }
    }

    // 12) out = x2 + moe, back to [B, C, T, T, T]
    {
        long total = (long)B_ * C_ * S_;
        final_combine<<<(unsigned)((total + 255) / 256), 256, 0, stream>>>(
            x2buf, moebuf, out);
    }
}
